// MultiSimilarityLoss_52029233824286
// MI455X (gfx1250) — matrix operand order repaired and checked by reading, not compiled, not run on hardware
//
#include <hip/hip_runtime.h>
#include <hip/hip_bf16.h>

typedef __attribute__((ext_vector_type(16))) __bf16 v16bf;
typedef __attribute__((ext_vector_type(8)))  float  v8f;

#define D_DIM 256
#define KCH   8      // 256 / 32 k-chunks
#define WAVES 8
#define BLOCK 256

static __device__ __forceinline__ unsigned short f32_to_bf16_rne(float x) {
    unsigned int u = __float_as_uint(x);
    u += 0x7FFFu + ((u >> 16) & 1u);
    return (unsigned short)(u >> 16);
}
static __device__ __forceinline__ float bf16_to_f32(unsigned short h) {
    return __uint_as_float(((unsigned int)h) << 16);
}

// ---------------------------------------------------------------------------
// Kernel 1: row-normalize feats, emit split-bf16 planes hi/lo. One wave / row.
// ---------------------------------------------------------------------------
__global__ __launch_bounds__(BLOCK) void msloss_normalize(
    const float* __restrict__ feats,
    unsigned short* __restrict__ hi,
    unsigned short* __restrict__ lo,
    float* __restrict__ out, int B) {
    if (blockIdx.x == 0 && threadIdx.x == 0) out[0] = 0.0f;
    const int wave = threadIdx.x >> 5;
    const int lane = threadIdx.x & 31;
    const int row  = blockIdx.x * 8 + wave;
    if (row >= B) return;

    const size_t base = (size_t)row * D_DIM + lane * 8;
    float4 a = *(const float4*)(feats + base);
    float4 b = *(const float4*)(feats + base + 4);
    float s = a.x*a.x + a.y*a.y + a.z*a.z + a.w*a.w
            + b.x*b.x + b.y*b.y + b.z*b.z + b.w*b.w;
    #pragma unroll
    for (int m = 16; m >= 1; m >>= 1) s += __shfl_xor(s, m, 32);
    const float inv = 1.0f / (sqrtf(s) + 1e-12f);

    float v[8] = {a.x, a.y, a.z, a.w, b.x, b.y, b.z, b.w};
    union { unsigned short s[8]; uint4 q; } ph, pl;
    #pragma unroll
    for (int i = 0; i < 8; ++i) {
        float x = v[i] * inv;
        unsigned short h = f32_to_bf16_rne(x);
        ph.s[i] = h;
        pl.s[i] = f32_to_bf16_rne(x - bf16_to_f32(h));
    }
    *(uint4*)(hi + base) = ph.q;
    *(uint4*)(lo + base) = pl.q;
}

// ---------------------------------------------------------------------------
// Kernel 2: fused Gram(WMMA split-bf16) + mining + loss. One block per 16 rows.
// ---------------------------------------------------------------------------
__global__ __launch_bounds__(BLOCK) void msloss_main(
    const unsigned short* __restrict__ hi,
    const unsigned short* __restrict__ lo,
    const int* __restrict__ labels,
    float* __restrict__ out, int B) {

    const int lane   = threadIdx.x & 31;
    const int wave   = threadIdx.x >> 5;
    const int half   = lane >> 4;     // 0: rows 0-7 / K-lo, 1: rows 8-15 / K-hi
    const int l15    = lane & 15;
    const int rowBase = blockIdx.x * 16;
    const int NT      = B / 16;       // N tiles

    __shared__ float s_a[WAVES][16], s_b[WAVES][16], s_c[WAVES][16], s_d[WAVES][16];
    __shared__ float s_minp[16], s_maxn[16], s_loss[16];

    // labels of the 8 rows this lane's accumulator covers (M = half*8 + r)
    int labi[8];
    #pragma unroll
    for (int r = 0; r < 8; ++r) labi[r] = labels[rowBase + half * 8 + r];

    // --- preload A fragments (hi & lo planes) for all 8 k-chunks -----------
    // A layout (16x32 bf16): lane half 0 -> K {0..7,16..23}, half 1 -> K {8..15,24..31}
    v16bf a_hi[KCH], a_lo[KCH];
    {
        const unsigned short* rp  = hi + (size_t)(rowBase + l15) * D_DIM;
        const unsigned short* rpl = lo + (size_t)(rowBase + l15) * D_DIM;
        const int aoff = half * 8;
        #pragma unroll
        for (int c = 0; c < KCH; ++c) {
            union { uint4 q[2]; v16bf v; } fh, fl;
            fh.q[0] = *(const uint4*)(rp  + c * 32 + aoff);
            fh.q[1] = *(const uint4*)(rp  + c * 32 + 16 + aoff);
            fl.q[0] = *(const uint4*)(rpl + c * 32 + aoff);
            fl.q[1] = *(const uint4*)(rpl + c * 32 + 16 + aoff);
            a_hi[c] = fh.v;
            a_lo[c] = fl.v;
        }
    }

    // ============================ SWEEP 1: mining ==========================
    float minp[8], maxn[8];
    #pragma unroll
    for (int r = 0; r < 8; ++r) { minp[r] = INFINITY; maxn[r] = -INFINITY; }

    for (int t = wave; t < NT; t += WAVES) {
        const int col  = t * 16 + l15;
        const int labj = labels[col];
        const int boff = half * 8;    // B layout: half 0 -> K {0..7,16..23}, half 1 -> K {8..15,24..31}
        const unsigned short* cph = hi + (size_t)col * D_DIM + boff;
        const unsigned short* cpl = lo + (size_t)col * D_DIM + boff;
        v8f acc = {};
        #pragma unroll
        for (int c = 0; c < KCH; ++c) {
            union { uint4 q[2]; v16bf v; } bh, bl;
            bh.q[0] = *(const uint4*)(cph + c * 32);
            bh.q[1] = *(const uint4*)(cph + c * 32 + 16);
            bl.q[0] = *(const uint4*)(cpl + c * 32);
            bl.q[1] = *(const uint4*)(cpl + c * 32 + 16);
            acc = __builtin_amdgcn_wmma_f32_16x16x32_bf16(false, a_hi[c], false, bh.v, (short)0, acc, false, false);
            acc = __builtin_amdgcn_wmma_f32_16x16x32_bf16(false, a_hi[c], false, bl.v, (short)0, acc, false, false);
            acc = __builtin_amdgcn_wmma_f32_16x16x32_bf16(false, a_lo[c], false, bh.v, (short)0, acc, false, false);
        }
        #pragma unroll
        for (int r = 0; r < 8; ++r) {
            float dist = sqrtf(fmaxf(2.0f - 2.0f * acc[r], 1e-12f));
            bool same = (labi[r] == labj);
            if (same && dist < (1.0f - 1e-5f)) minp[r] = fminf(minp[r], dist);
            if (!same)                         maxn[r] = fmaxf(maxn[r], dist);
        }
    }
    #pragma unroll
    for (int m = 8; m >= 1; m >>= 1)
        #pragma unroll
        for (int r = 0; r < 8; ++r) {
            minp[r] = fminf(minp[r], __shfl_xor(minp[r], m, 32));
            maxn[r] = fmaxf(maxn[r], __shfl_xor(maxn[r], m, 32));
        }
    if (l15 == 0)
        #pragma unroll
        for (int r = 0; r < 8; ++r) {
            s_a[wave][half * 8 + r] = minp[r];
            s_b[wave][half * 8 + r] = maxn[r];
        }
    __syncthreads();
    if (threadIdx.x < 16) {
        float mn = INFINITY, mx = -INFINITY;
        for (int w = 0; w < WAVES; ++w) {
            mn = fminf(mn, s_a[w][threadIdx.x]);
            mx = fmaxf(mx, s_b[w][threadIdx.x]);
        }
        s_minp[threadIdx.x] = mn;
        s_maxn[threadIdx.x] = mx;
    }
    __syncthreads();

    float thr_min[8], thr_max[8];
    #pragma unroll
    for (int r = 0; r < 8; ++r) {
        thr_min[r] = s_minp[half * 8 + r];
        thr_max[r] = s_maxn[half * 8 + r];
    }
    __syncthreads();

    // ====================== SWEEP 2: masked exp-sums =======================
    float nsum[8], psum[8], ncnt[8], pcnt[8];
    #pragma unroll
    for (int r = 0; r < 8; ++r) { nsum[r] = psum[r] = ncnt[r] = pcnt[r] = 0.0f; }

    for (int t = wave; t < NT; t += WAVES) {
        const int col  = t * 16 + l15;
        const int labj = labels[col];
        const int boff = half * 8;
        const unsigned short* cph = hi + (size_t)col * D_DIM + boff;
        const unsigned short* cpl = lo + (size_t)col * D_DIM + boff;
        v8f acc = {};
        #pragma unroll
        for (int c = 0; c < KCH; ++c) {
            union { uint4 q[2]; v16bf v; } bh, bl;
            bh.q[0] = *(const uint4*)(cph + c * 32);
            bh.q[1] = *(const uint4*)(cph + c * 32 + 16);
            bl.q[0] = *(const uint4*)(cpl + c * 32);
            bl.q[1] = *(const uint4*)(cpl + c * 32 + 16);
            acc = __builtin_amdgcn_wmma_f32_16x16x32_bf16(false, a_hi[c], false, bh.v, (short)0, acc, false, false);
            acc = __builtin_amdgcn_wmma_f32_16x16x32_bf16(false, a_hi[c], false, bl.v, (short)0, acc, false, false);
            acc = __builtin_amdgcn_wmma_f32_16x16x32_bf16(false, a_lo[c], false, bh.v, (short)0, acc, false, false);
        }
        #pragma unroll
        for (int r = 0; r < 8; ++r) {
            float dist = sqrtf(fmaxf(2.0f - 2.0f * acc[r], 1e-12f));
            bool same = (labi[r] == labj);
            bool nm = !same && (dist + 0.1f > thr_min[r]);
            bool pm = same && (dist < (1.0f - 1e-5f)) && (dist - 0.1f < thr_max[r]);
            if (nm) { nsum[r] += expf(-40.0f * (dist - 1.0f)); ncnt[r] += 1.0f; }
            if (pm) { psum[r] += expf(  2.0f * (dist - 0.7f)); pcnt[r] += 1.0f; }
        }
    }
    #pragma unroll
    for (int m = 8; m >= 1; m >>= 1)
        #pragma unroll
        for (int r = 0; r < 8; ++r) {
            nsum[r] += __shfl_xor(nsum[r], m, 32);
            psum[r] += __shfl_xor(psum[r], m, 32);
            ncnt[r] += __shfl_xor(ncnt[r], m, 32);
            pcnt[r] += __shfl_xor(pcnt[r], m, 32);
        }
    if (l15 == 0)
        #pragma unroll
        for (int r = 0; r < 8; ++r) {
            s_a[wave][half * 8 + r] = nsum[r];
            s_b[wave][half * 8 + r] = psum[r];
            s_c[wave][half * 8 + r] = ncnt[r];
            s_d[wave][half * 8 + r] = pcnt[r];
        }
    __syncthreads();
    if (threadIdx.x < 16) {
        float ns = 0.f, ps = 0.f, nc = 0.f, pc = 0.f;
        for (int w = 0; w < WAVES; ++w) {
            ns += s_a[w][threadIdx.x]; ps += s_b[w][threadIdx.x];
            nc += s_c[w][threadIdx.x]; pc += s_d[w][threadIdx.x];
        }
        float neg_loss = log1pf(ns) * (1.0f / 40.0f);
        float pos_loss = log1pf(ps) / (pc + 1e-5f);
        s_loss[threadIdx.x] = (nc >= 1.0f && pc >= 1.0f) ? (pos_loss + neg_loss) : 0.0f;
    }
    __syncthreads();
    if (threadIdx.x == 0) {
        float tot = 0.f;
        #pragma unroll
        for (int r = 0; r < 16; ++r) tot += s_loss[r];
        atomicAdd(out, tot / (float)B);
    }
}

extern "C" void kernel_launch(void* const* d_in, const int* in_sizes, int n_in,
                              void* d_out, int out_size, void* d_ws, size_t ws_size,
                              hipStream_t stream) {
    const float* feats  = (const float*)d_in[0];
    const int*   labels = (const int*)d_in[1];
    float*       out    = (float*)d_out;
    const int B = in_sizes[1];                       // 8192

    unsigned short* hi = (unsigned short*)d_ws;      // B*256 bf16 (4 MB)
    unsigned short* lo = hi + (size_t)B * D_DIM;     // B*256 bf16 (4 MB)

    msloss_normalize<<<(B + 7) / 8, BLOCK, 0, stream>>>(feats, hi, lo, out, B);
    msloss_main<<<B / 16, BLOCK, 0, stream>>>(hi, lo, labels, out, B);
}
